// GaussianBasis_68994354643524
// MI455X (gfx1250) — matrix operand order repaired and checked by reading, not compiled, not run on hardware
//
#include <hip/hip_runtime.h>
#include <hip/hip_bf16.h>

typedef __attribute__((ext_vector_type(16))) _Float16 v16h;
typedef __attribute__((ext_vector_type(8)))  float    v8f;
typedef __attribute__((ext_vector_type(4)))  unsigned int u4v;
typedef __attribute__((ext_vector_type(4)))  float    f4v;

#define NGAUSS 1024
#define IMG    256
#define HW     (IMG * IMG)
#define MROWS  48

union AFrag { u4v u[2]; v16h h; };

// ---------------------------------------------------------------------------
// Pre-pass: repack features (C=16, N=1024, K=3) f32 -> f16 A matrix [m][n],
// m = c*3+k, row-major with stride NGAUSS. 48*1024*2B = 96 KB in d_ws.
// ---------------------------------------------------------------------------
__global__ __launch_bounds__(256) void pack_A_kernel(
    const float* __restrict__ feat, _Float16* __restrict__ A)
{
    int idx = blockIdx.x * 256 + threadIdx.x;   // 0 .. 49151
    int m = idx >> 10;                          // / 1024
    int n = idx & (NGAUSS - 1);
    int c = m / 3;
    int k = m - 3 * c;
    A[idx] = (_Float16)feat[(c * NGAUSS + n) * 3 + k];
}

// ---------------------------------------------------------------------------
// Fused splat + GEMM. One block per image row (y = blockIdx.x).
// All y-dependent per-Gaussian terms are hoisted into the prologue, with
// -log2(e) pre-folded so the inner loop is: 2 FMAs -> v_exp_f32 -> mul/min/sel.
// 8 waves; wave w owns pixel tiles [w*32, w*32+16) and [w*32+16, w*32+32).
// GEMM: D(48 x 256-row-slice) += A(48 x 1024) * B(1024 x 256), B generated.
// ---------------------------------------------------------------------------
__global__ __launch_bounds__(256) void splat_gemm_kernel(
    const float* __restrict__ xyz_raw,
    const float* __restrict__ chol_raw,
    const float* __restrict__ opac,
    const _Float16* __restrict__ A,
    float* __restrict__ out)
{
    // Per-Gaussian row-specialized params, 8-float stride (b128-aligned): 32 KB
    __shared__ float sp[NGAUSS * 8];

    const int tid = threadIdx.x;
    const int y   = blockIdx.x;
    const float py = (float)y + 0.5f;

    // ---- prologue: per-(Gaussian, row) derived params (each thread does 4) ----
    for (int g = tid; g < NGAUSS; g += 256) {
        float mx  = tanhf(xyz_raw[2 * g + 0]);
        float my  = tanhf(xyz_raw[2 * g + 1]);
        float xyX = 0.5f * (mx + 1.0f) * (float)IMG;   // W = 256
        float xyY = 0.5f * (my + 1.0f) * (float)IMG;   // H = 256
        float l1  = chol_raw[3 * g + 0] + 0.5f;
        float l2  = chol_raw[3 * g + 1];
        float l3  = chol_raw[3 * g + 2] + 0.5f;
        float sxx = l1 * l1;
        float sxy = l1 * l2;
        float syy = l2 * l2 + l3 * l3;
        float inv = 1.0f / (sxx * syy - sxy * sxy);
        float a   = 0.5f * syy * inv;    // 0.5*cxx
        float c   = -sxy * inv;          // cxy
        float b   = 0.5f * sxx * inv;    // 0.5*cyy

        const float L2E = 1.4426950408889634f;  // log2(e)
        float dy   = py - xyY;
        float na   = -a * L2E;                  // s = -(sigma)*log2e
        float ncdy = -(c * dy) * L2E;
        float nbyy = -(b * dy * dy) * L2E;

        float* p = &sp[g * 8];
        p[0] = xyX;
        p[1] = na;
        p[2] = ncdy;
        p[3] = nbyy;
        p[4] = opac[g];
        p[5] = 0.0f;
        p[6] = 0.0f;
        p[7] = 0.0f;
    }
    __syncthreads();

    const int lane  = tid & 31;
    const int wave  = tid >> 5;
    const int laneM = lane & 15;   // row within A tile / column within B,C tile
    const int hf    = lane >> 4;   // which K/M half this lane handles

    const float px0 = (float)(wave * 32 + laneM) + 0.5f;   // tile 0 pixel x

    v8f acc[3][2];
#pragma unroll
    for (int mt = 0; mt < 3; ++mt) {
        v8f z = {};
        acc[mt][0] = z;
        acc[mt][1] = z;
    }

    for (int k0 = 0; k0 < NGAUSS; k0 += 32) {
        // ---- build B fragments (alpha values) in WMMA layout ----
        // lane half hf: elements 0..7 hold K = k0+hf*8..+7, elements 8..15 hold K = k0+16+hf*8..+7
        v16h b0, b1;
        const int gbase = k0 + hf * 8;
#pragma unroll
        for (int j = 0; j < 16; ++j) {
            const float* p = &sp[(gbase + j + (j >= 8 ? 8 : 0)) * 8];
            f4v pv   = *(const f4v*)p;   // xyX, na, ncdy, nbyy
            float op = p[4];

            float dx0 = px0 - pv.x;
            float s0  = __fmaf_rn(dx0, __fmaf_rn(pv.y, dx0, pv.z), pv.w);
            float e0  = op * __builtin_amdgcn_exp2f(s0);   // v_exp_f32
            float a0  = (s0 <= 0.0f) ? fminf(e0, 0.999f) : 0.0f;

            float dx1 = dx0 + 16.0f;                       // tile-1 pixel is +16 in x
            float s1  = __fmaf_rn(dx1, __fmaf_rn(pv.y, dx1, pv.z), pv.w);
            float e1  = op * __builtin_amdgcn_exp2f(s1);
            float a1  = (s1 <= 0.0f) ? fminf(e1, 0.999f) : 0.0f;

            b0[j] = (_Float16)a0;
            b1[j] = (_Float16)a1;
        }

        // ---- load A fragments (3 M-tiles) and issue WMMAs ----
#pragma unroll
        for (int mt = 0; mt < 3; ++mt) {
            const int row = mt * 16 + laneM;
            const _Float16* ap = A + row * NGAUSS + k0 + hf * 8;
            AFrag af;
            af.u[0] = *(const u4v*)ap;         // K = k0 + hf*8 .. +7
            af.u[1] = *(const u4v*)(ap + 16);  // K = k0 + 16 + hf*8 .. +7
            acc[mt][0] = __builtin_amdgcn_wmma_f32_16x16x32_f16(
                false, af.h, false, b0, (short)0, acc[mt][0], false, false);
            acc[mt][1] = __builtin_amdgcn_wmma_f32_16x16x32_f16(
                false, af.h, false, b1, (short)0, acc[mt][1], false, false);
        }
    }

    // ---- epilogue: C layout -> VGPR r holds row (hf*8 + r), col laneM ----
#pragma unroll
    for (int mt = 0; mt < 3; ++mt) {
#pragma unroll
        for (int nt = 0; nt < 2; ++nt) {
            const int x = wave * 32 + nt * 16 + laneM;
            const long p = (long)y * IMG + x;
#pragma unroll
            for (int r = 0; r < 8; ++r) {
                const int m = mt * 16 + hf * 8 + r;
                out[(long)m * HW + p] = acc[mt][nt][r];
            }
        }
    }
}

extern "C" void kernel_launch(void* const* d_in, const int* in_sizes, int n_in,
                              void* d_out, int out_size, void* d_ws, size_t ws_size,
                              hipStream_t stream) {
    (void)in_sizes; (void)n_in; (void)out_size; (void)ws_size;
    const float* xyz  = (const float*)d_in[0];   // (N,2)
    const float* chol = (const float*)d_in[1];   // (N,3)
    const float* feat = (const float*)d_in[2];   // (C,N,3)
    const float* opac = (const float*)d_in[3];   // (N,1)
    float* out = (float*)d_out;                  // (C,3,H,W) = 48*65536 f32
    _Float16* A = (_Float16*)d_ws;               // 96 KB staging

    pack_A_kernel<<<(MROWS * NGAUSS) / 256, 256, 0, stream>>>(feat, A);
    splat_gemm_kernel<<<IMG, 256, 0, stream>>>(xyz, chol, opac, A, out);
}
